// LocalAttention_37374805409903
// MI455X (gfx1250) — matrix operand order repaired and checked by reading, not compiled, not run on hardware
//
#include <hip/hip_runtime.h>

// ---------------- problem constants ----------------
#define S_LEN   2048
#define HID     1024
#define NHEAD   16
#define HDIM    64
#define WIN     256
#define SCALE_F 0.125f   // 64^-0.5

typedef __attribute__((ext_vector_type(16))) _Float16 v16h;
typedef __attribute__((ext_vector_type(8)))  _Float16 v8h;
typedef __attribute__((ext_vector_type(8)))  float    v8f;

__device__ __forceinline__ v8f wmma_f16(v16h a, v16h b, v8f c) {
  // D = A(16x32 f16) * B(32x16 f16) + C(16x16 f32)
  return __builtin_amdgcn_wmma_f32_16x16x32_f16(false, a, false, b, (short)0, c,
                                                false, false);
}

// ---- A fragment (16x32 f16) from row-major src, stride in halves ----
// lane<16: row=lane, elems 0..7 = K 0..7, 8..15 = K 16..23
// lane>=16: row=lane-16, elems 0..7 = K 8..15, 8..15 = K 24..31
__device__ __forceinline__ v16h load_a_frag(const _Float16* __restrict__ src, int stride) {
  int lane = threadIdx.x & 31;
  int row  = lane & 15;
  int kb   = (lane < 16) ? 0 : 8;
  const _Float16* p = src + row * stride + kb;
  v8h lo = *(const v8h*)(p);
  v8h hi = *(const v8h*)(p + 16);
  v16h a;
#pragma unroll
  for (int e = 0; e < 8; ++e) { a[e] = lo[e]; a[8 + e] = hi[e]; }
  return a;
}

// ---- B fragment (32x16) where per-column data is CONTIGUOUS in memory ----
// element(K=kb+e, N=col) = base[col*colstride + koff + kb + e]
// (used for W^T [N][K], K^T via Kh[s][d], and V^T [d][s])
__device__ __forceinline__ v16h load_bT_frag(const _Float16* __restrict__ base,
                                             int colstride, int koff) {
  int lane = threadIdx.x & 31;
  int col  = lane & 15;
  int kb   = (lane < 16) ? 0 : 8;
  const _Float16* p = base + (size_t)col * colstride + koff + kb;
  v8h lo = *(const v8h*)(p);
  v8h hi = *(const v8h*)(p + 16);
  v16h b;
#pragma unroll
  for (int e = 0; e < 8; ++e) { b[e] = lo[e]; b[8 + e] = hi[e]; }
  return b;
}

// =============== fp32 -> f16 conversion ===============
__global__ void cvt_f32_to_f16(const float* __restrict__ src, _Float16* __restrict__ dst, int n) {
  int i = blockIdx.x * blockDim.x + threadIdx.x;
  int stride = gridDim.x * blockDim.x;
  for (; i < n; i += stride) dst[i] = (_Float16)src[i];
}

// =============== fp32 [K][N] -> f16 [N][K] tiled transpose ===============
// block (32,8), grid (N/32, K/32); both dims = 1024
__global__ void transpose_cvt(const float* __restrict__ src, _Float16* __restrict__ dst) {
  __shared__ _Float16 tile[32][34];
  int bx = blockIdx.x * 32;   // N base
  int by = blockIdx.y * 32;   // K base
  int tx = threadIdx.x, ty = threadIdx.y;
#pragma unroll
  for (int i = 0; i < 32; i += 8)
    tile[ty + i][tx] = (_Float16)src[(size_t)(by + ty + i) * HID + bx + tx];
  __syncthreads();
#pragma unroll
  for (int i = 0; i < 32; i += 8)
    dst[(size_t)(bx + ty + i) * HID + by + tx] = tile[tx][ty + i];
}

// =============== fused QKV projection ===============
// x16:[2048,1024] f16; w*T:[N=1024][K=1024] f16 (transposed)
// out: Q/K in [head][seq][64] f16, V in [head][64][seq] f16 (transposed)
// grid=(48,8), block=256 (8 waves); wave tile = 32(M) x 64(N)
__global__ void qkv_gemm(const _Float16* __restrict__ x16,
                         const _Float16* __restrict__ wqT,
                         const _Float16* __restrict__ wkT,
                         const _Float16* __restrict__ wvT,
                         const float* __restrict__ bq,
                         const float* __restrict__ bk,
                         const float* __restrict__ bv,
                         _Float16* __restrict__ Qh,
                         _Float16* __restrict__ Kh,
                         _Float16* __restrict__ Vt) {
  int tid  = threadIdx.x;
  int wave = tid >> 5;
  int lane = tid & 31;

  int n0  = blockIdx.x * 64;                 // 0..3071
  int sel = n0 >> 10;                        // 0:Q 1:K 2:V
  int within = n0 & 1023;                    // col in the selected matrix
  const _Float16* wT  = (sel == 0) ? wqT : (sel == 1) ? wkT : wvT;
  const float*    bias = (sel == 0) ? bq  : (sel == 1) ? bk  : bv;

  int m0 = blockIdx.y * 256 + wave * 32;     // 32 rows per wave

  v8f acc[2][4];
#pragma unroll
  for (int h2 = 0; h2 < 2; ++h2)
#pragma unroll
    for (int j = 0; j < 4; ++j) acc[h2][j] = (v8f)(0.f);

  for (int kk = 0; kk < HID; kk += 32) {
    v16h a0 = load_a_frag(x16 + (size_t)m0 * HID + kk, HID);
    v16h a1 = load_a_frag(x16 + (size_t)(m0 + 16) * HID + kk, HID);
#pragma unroll
    for (int j = 0; j < 4; ++j) {
      v16h b = load_bT_frag(wT + (size_t)(within + j * 16) * HID, HID, kk);
      acc[0][j] = wmma_f16(a0, b, acc[0][j]);
      acc[1][j] = wmma_f16(a1, b, acc[1][j]);
    }
  }

  // epilogue
  int head = within >> 6;
  int colN = lane & 15;
  int rowAdd = (lane >= 16) ? 8 : 0;

  if (sel == 2) {
    // V transposed: Vt[head][d][s]
    _Float16* dh = Vt + (size_t)head * HDIM * S_LEN;
#pragma unroll
    for (int h2 = 0; h2 < 2; ++h2) {
      int sbase = m0 + h2 * 16 + rowAdd;
#pragma unroll
      for (int j = 0; j < 4; ++j) {
        int dcol = j * 16 + colN;
        float bb = bias[within + dcol];
        v8h pack;
#pragma unroll
        for (int r = 0; r < 8; ++r) pack[r] = (_Float16)(acc[h2][j][r] + bb);
        *(v8h*)(dh + (size_t)dcol * S_LEN + sbase) = pack;   // 8 contiguous s
      }
    }
  } else {
    _Float16* dst = (sel == 0) ? Qh : Kh;
    _Float16* dh = dst + (size_t)head * S_LEN * HDIM;        // [s][64]
#pragma unroll
    for (int h2 = 0; h2 < 2; ++h2) {
#pragma unroll
      for (int j = 0; j < 4; ++j) {
        int dcol = j * 16 + colN;
        float bb = bias[within + dcol];
#pragma unroll
        for (int r = 0; r < 8; ++r) {
          int s = m0 + h2 * 16 + r + rowAdd;
          dh[(size_t)s * HDIM + dcol] = (_Float16)(acc[h2][j][r] + bb);
        }
      }
    }
  }
}

// =============== windowed flash attention ===============
// one wave = one (head, 16-query tile). grid=(2,128), block=256 (8 waves)
// all waves in a block share q0 -> identical control flow.
__global__ void attn_kernel(const _Float16* __restrict__ Qh,
                            const _Float16* __restrict__ Kh,
                            const _Float16* __restrict__ Vt,
                            _Float16* __restrict__ Ah) {
  __shared__ _Float16 Pbuf[8][16 * 32];     // per-wave P tile (16q x 32k)

  int wave = threadIdx.x >> 5;
  int lane = threadIdx.x & 31;
  int head = blockIdx.x * 8 + wave;         // 0..15
  int q0   = blockIdx.y * 16;               // query tile start

  const _Float16* qhead  = Qh + (size_t)head * S_LEN * HDIM;   // [s][64]
  const _Float16* khead  = Kh + (size_t)head * S_LEN * HDIM;   // [s][64]
  const _Float16* vthead = Vt + (size_t)head * HDIM * S_LEN;   // [d][s]

  // Q fragments over d (K-dim 64 = 2 x 32)
  v16h qa0 = load_a_frag(qhead + (size_t)q0 * HDIM, HDIM);
  v16h qa1 = load_a_frag(qhead + (size_t)q0 * HDIM + 32, HDIM);

  v8f o[4];
#pragma unroll
  for (int j = 0; j < 4; ++j) o[j] = (v8f)(0.f);
  float m[8], l[8];
#pragma unroll
  for (int r = 0; r < 8; ++r) { m[r] = -1e30f; l[r] = 0.f; }

  int colN = lane & 15;
  int rowAdd = (lane >= 16) ? 8 : 0;

  int kfirst = q0 - (WIN - 1); if (kfirst < 0) kfirst = 0;
  int kbeg = kfirst & ~31;                  // 32-aligned => kc <= 2016, all reads in-bounds
  int kend = q0 + 15;

  for (int kc = kbeg; kc <= kend; kc += 32) {
    // ---- scores: two 16x16 tiles over keys [kc,kc+15], [kc+16,kc+31] ----
    // K^T fragment: element(K=d, N=key) = khead[key*64 + d]  (contiguous in d)
    v8f s0 = (v8f)(0.f), s1 = (v8f)(0.f);
    s0 = wmma_f16(qa0, load_bT_frag(khead + (size_t)kc * HDIM, HDIM, 0), s0);
    s0 = wmma_f16(qa1, load_bT_frag(khead + (size_t)kc * HDIM, HDIM, 32), s0);
    s1 = wmma_f16(qa0, load_bT_frag(khead + (size_t)(kc + 16) * HDIM, HDIM, 0), s1);
    s1 = wmma_f16(qa1, load_bT_frag(khead + (size_t)(kc + 16) * HDIM, HDIM, 32), s1);

    // ---- mask + scale ----
    float sc0[8], sc1[8];
    bool ok0[8], ok1[8];
#pragma unroll
    for (int r = 0; r < 8; ++r) {
      int q = q0 + r + rowAdd;
      int key0 = kc + colN;
      int key1 = kc + 16 + colN;
      ok0[r] = (key0 <= q) && (key0 > q - WIN);
      ok1[r] = (key1 <= q) && (key1 > q - WIN);
      sc0[r] = ok0[r] ? s0[r] * SCALE_F : -1e30f;
      sc1[r] = ok1[r] ? s1[r] * SCALE_F : -1e30f;
    }

    // ---- online softmax (row groups = 16 lanes within each half-wave) ----
    float alph[8], p0s[8], p1s[8];
#pragma unroll
    for (int r = 0; r < 8; ++r) {
      float v = fmaxf(sc0[r], sc1[r]);
      v = fmaxf(v, __shfl_xor(v, 1));
      v = fmaxf(v, __shfl_xor(v, 2));
      v = fmaxf(v, __shfl_xor(v, 4));
      v = fmaxf(v, __shfl_xor(v, 8));
      float mnew = fmaxf(m[r], v);
      float a = __expf(m[r] - mnew);
      m[r] = mnew;
      float p0 = ok0[r] ? __expf(sc0[r] - mnew) : 0.f;   // explicit zero: safe
      float p1 = ok1[r] ? __expf(sc1[r] - mnew) : 0.f;   // for all-masked rows
      float ps = p0 + p1;
      ps += __shfl_xor(ps, 1);
      ps += __shfl_xor(ps, 2);
      ps += __shfl_xor(ps, 4);
      ps += __shfl_xor(ps, 8);
      l[r] = l[r] * a + ps;
      alph[r] = a; p0s[r] = p0; p1s[r] = p1;
    }

    // ---- rescale accumulators ----
#pragma unroll
    for (int j = 0; j < 4; ++j)
#pragma unroll
      for (int r = 0; r < 8; ++r) o[j][r] *= alph[r];

    // ---- P (C-layout) -> LDS -> A-layout fragment ----
    _Float16* P = &Pbuf[wave][0];
#pragma unroll
    for (int r = 0; r < 8; ++r) {
      P[(r + rowAdd) * 32 + colN]      = (_Float16)p0s[r];
      P[(r + rowAdd) * 32 + 16 + colN] = (_Float16)p1s[r];
    }
    asm volatile("s_wait_dscnt 0x0" ::: "memory");   // LDS in-order per wave; block reordering
    v16h pa = load_a_frag(P, 32);

    // ---- PV: 4 d-tiles of 16; V^T fragment is contiguous along s ----
#pragma unroll
    for (int j = 0; j < 4; ++j) {
      v16h vb = load_bT_frag(vthead + (size_t)(j * 16) * S_LEN, S_LEN, kc);
      o[j] = wmma_f16(pa, vb, o[j]);
    }
    asm volatile("" ::: "memory");   // keep P stores of next iter after the loads
  }

  // ---- normalize and store to Ah[s][h*64+d] f16 ----
#pragma unroll
  for (int j = 0; j < 4; ++j) {
#pragma unroll
    for (int r = 0; r < 8; ++r) {
      int s = q0 + r + rowAdd;
      float ov = o[j][r] / l[r];
      Ah[(size_t)s * HID + head * HDIM + j * 16 + colN] = (_Float16)ov;
    }
  }
}

// =============== output projection (fp32 out) ===============
// grid=(16,8), block=256 (8 waves); wave tile = 32(M) x 64(N)
__global__ void out_gemm(const _Float16* __restrict__ Ah,
                         const _Float16* __restrict__ woT,
                         const float* __restrict__ bo,
                         float* __restrict__ out) {
  int tid  = threadIdx.x;
  int wave = tid >> 5;
  int lane = tid & 31;
  int n0 = blockIdx.x * 64;
  int m0 = blockIdx.y * 256 + wave * 32;

  v8f acc[2][4];
#pragma unroll
  for (int h2 = 0; h2 < 2; ++h2)
#pragma unroll
    for (int j = 0; j < 4; ++j) acc[h2][j] = (v8f)(0.f);

  for (int kk = 0; kk < HID; kk += 32) {
    v16h a0 = load_a_frag(Ah + (size_t)m0 * HID + kk, HID);
    v16h a1 = load_a_frag(Ah + (size_t)(m0 + 16) * HID + kk, HID);
#pragma unroll
    for (int j = 0; j < 4; ++j) {
      v16h b = load_bT_frag(woT + (size_t)(n0 + j * 16) * HID, HID, kk);
      acc[0][j] = wmma_f16(a0, b, acc[0][j]);
      acc[1][j] = wmma_f16(a1, b, acc[1][j]);
    }
  }

  int colN = lane & 15;
  int rowAdd = (lane >= 16) ? 8 : 0;
#pragma unroll
  for (int h2 = 0; h2 < 2; ++h2) {
#pragma unroll
    for (int j = 0; j < 4; ++j) {
      int n = n0 + j * 16 + colN;
      float bb = bo[n];
#pragma unroll
      for (int r = 0; r < 8; ++r) {
        int s = m0 + h2 * 16 + r + rowAdd;
        out[(size_t)s * HID + n] = acc[h2][j][r] + bb;
      }
    }
  }
}

// =============== host launch ===============
extern "C" void kernel_launch(void* const* d_in, const int* in_sizes, int n_in,
                              void* d_out, int out_size, void* d_ws, size_t ws_size,
                              hipStream_t stream) {
  const float* x  = (const float*)d_in[0];
  const float* wq = (const float*)d_in[1];
  const float* bq = (const float*)d_in[2];
  const float* wk = (const float*)d_in[3];
  const float* bk = (const float*)d_in[4];
  const float* wv = (const float*)d_in[5];
  const float* bv = (const float*)d_in[6];
  const float* wo = (const float*)d_in[7];
  const float* bo = (const float*)d_in[8];
  float* out = (float*)d_out;

  // workspace layout (halves)
  _Float16* ws = (_Float16*)d_ws;
  const size_t N_X = (size_t)S_LEN * HID;        // 2M
  const size_t N_W = (size_t)HID * HID;          // 1M
  _Float16* xh  = ws;
  _Float16* wqT = xh  + N_X;                     // [N][K]
  _Float16* wkT = wqT + N_W;
  _Float16* wvT = wkT + N_W;
  _Float16* woT = wvT + N_W;
  _Float16* Qh  = woT + N_W;                     // [16][2048][64]
  _Float16* Kh  = Qh + N_X;                      // [16][2048][64]
  _Float16* Vt  = Kh + N_X;                      // [16][64][2048]
  _Float16* Ah  = Vt + N_X;                      // [2048][1024]

  cvt_f32_to_f16<<<1024, 256, 0, stream>>>(x, xh, (int)N_X);
  dim3 tb(32, 8), tg(HID / 32, HID / 32);
  transpose_cvt<<<tg, tb, 0, stream>>>(wq, wqT);
  transpose_cvt<<<tg, tb, 0, stream>>>(wk, wkT);
  transpose_cvt<<<tg, tb, 0, stream>>>(wv, wvT);
  transpose_cvt<<<tg, tb, 0, stream>>>(wo, woT);

  qkv_gemm<<<dim3(48, 8), 256, 0, stream>>>(xh, wqT, wkT, wvT, bq, bk, bv, Qh, Kh, Vt);
  attn_kernel<<<dim3(2, 128), 256, 0, stream>>>(Qh, Kh, Vt, Ah);
  out_gemm<<<dim3(16, 8), 256, 0, stream>>>(Ah, woT, bo, out);
}
